// PWC_tf_13606456394322
// MI455X (gfx1250) — matrix operand order repaired and checked by reading, not compiled, not run on hardware
//
#include <hip/hip_runtime.h>

// ---------------------------------------------------------------------------
// PWC-Net forward for MI455X (gfx1250).
// Convs: implicit-GEMM on the WMMA pipe, bf16 inputs / f32 accumulate via
// v_wmma_f32_16x16x32_bf16.  A block owns SPB 16-pixel spatial tiles and all
// output-channel tiles; the 32x(16*SPB) im2col B fragment is gathered once per
// K-chunk by the whole block into LDS with branch-free clamped loads and an
// incremental (division-free) k -> (ic,ky,kx) decode, then each wave reads its
// lane fragment as 32 contiguous LDS bytes and feeds the WMMA.  EXEC stays
// all-ones through the WMMA (no divergent masking of the matrix op); out-of-
// range rows/columns compute garbage that the masked epilogue never stores.
// ---------------------------------------------------------------------------

typedef __attribute__((ext_vector_type(16))) __bf16 v16bf;
typedef __attribute__((ext_vector_type(8)))  float  v8f;

__device__ __forceinline__ __bf16 f2bf(float f) {
  unsigned int u = __builtin_bit_cast(unsigned int, f);
  unsigned int r = (u + 0x7FFFu + ((u >> 16) & 1u)) >> 16;   // round-nearest-even
  return __builtin_bit_cast(__bf16, (unsigned short)r);
}

// ---------------------------------------------------------------------------
// SPB = spatial 16-pixel sub-tiles per block (1,2,4,8); oc tiles = 8/SPB.
// grid.x = ceil(B*H*W / (16*SPB)); block = 256 threads (8 waves);
// wave w -> (sp_sub = w % SPB, oc_tile = w / SPB).
// ---------------------------------------------------------------------------
template <int SPB>
__global__ __launch_bounds__(256) void conv3x3_wmma_bf16(
    const float* __restrict__ in, const float* __restrict__ wgt,
    const float* __restrict__ bias, const float* __restrict__ res,
    float* __restrict__ out,
    int NB, int Cin, int H, int W, int Cout, int dil, int leaky)
{
  const int HW  = H * W;
  const int BHW = NB * HW;
  const int K   = Cin * 9;

  // Double-buffered B fragment: [buf][column][K within chunk], bf16.
  __shared__ __attribute__((aligned(64))) __bf16 ldsB[2][16 * SPB][32];

  const int tid      = threadIdx.x;
  const int lane     = tid & 31;
  const int wave     = tid >> 5;
  const int blk_base = blockIdx.x * (16 * SPB);

  // ---- staging role: thread stages K-pair kl for SPB columns --------------
  const int n_s = tid >> 4;                   // 0..15 (column within sub-tile)
  const int kl  = (tid & 15) << 1;            // 0,2,...,30
  int sbase[SPB];                             // &in[b*Cin*HW] offset per column
  int ys[SPB], xs[SPB];
  #pragma unroll
  for (int sub = 0; sub < SPB; ++sub) {
    const int s  = blk_base + sub * 16 + n_s;
    const int sv = (s < BHW) ? s : (BHW - 1);
    const int b  = sv / HW;
    const int r  = sv - b * HW;
    ys[sub] = r / W;
    xs[sub] = r - ys[sub] * W;
    sbase[sub] = b * Cin * HW;
  }
  // incremental k -> (ic, rr) state for the two staged elements (k = k0+kl+t)
  int ic0 = kl / 9,       rr0 = kl - 9 * ic0;
  int ic1 = (kl + 1) / 9, rr1 = (kl + 1) - 9 * ic1;

  // ---- compute role -------------------------------------------------------
  const int sp_sub  = wave % SPB;             // wave-uniform
  const int oc_t    = wave / SPB;
  const int oc_base = oc_t << 4;
  const int colb    = sp_sub * 16;            // LDS column base for this wave
  const int n       = lane & 15;
  const int half    = lane >> 4;              // 0: K 0..15, 1: K 16..31
  const int s       = blk_base + colb + n;
  const int sv      = (s < BHW) ? s : (BHW - 1);
  const int b       = sv / HW;
  const int r2      = sv - b * HW;
  const int y       = r2 / W;
  const int x       = r2 - y * W;

  const int arow  = oc_base + n;
  const int arowc = (arow < Cout) ? arow : (Cout - 1);   // address clamp only
  const float* wrow = wgt + (long long)arowc * K;
  const int khiA = half ? 16 : 0;             // A: lanes 16-31 hold K+16..K+31

  v8f acc = {};
  int buf = 0;
  for (int k0 = 0; k0 < K; k0 += 32, buf ^= 1) {
    // ---- division-free tap decode for the two staged K values ------------
    const int ky0 = rr0 >= 6 ? 2 : (rr0 >= 3 ? 1 : 0);
    const int ky1 = rr1 >= 6 ? 2 : (rr1 >= 3 ? 1 : 0);
    const int dy0 = (ky0 - 1) * dil, dx0 = (rr0 - 3 * ky0 - 1) * dil;
    const int dy1 = (ky1 - 1) * dil, dx1 = (rr1 - 3 * ky1 - 1) * dil;
    const bool kok0 = (k0 + kl)     < K;
    const bool kok1 = (k0 + kl + 1) < K;
    const int icc0 = kok0 ? ic0 : (Cin - 1);  // address clamp for K tail
    const int icc1 = kok1 ? ic1 : (Cin - 1);

    #pragma unroll
    for (int sub = 0; sub < SPB; ++sub) {
      unsigned int pack = 0;
      {
        const int iy = ys[sub] + dy0, ix = xs[sub] + dx0;
        const int iyc = iy < 0 ? 0 : (iy >= H ? H - 1 : iy);
        const int ixc = ix < 0 ? 0 : (ix >= W ? W - 1 : ix);
        const float v = in[sbase[sub] + icc0 * HW + iyc * W + ixc];
        const bool ok = kok0 && (iy == iyc) && (ix == ixc);
        pack = (unsigned int)__builtin_bit_cast(unsigned short,
                                                f2bf(ok ? v : 0.f));
      }
      {
        const int iy = ys[sub] + dy1, ix = xs[sub] + dx1;
        const int iyc = iy < 0 ? 0 : (iy >= H ? H - 1 : iy);
        const int ixc = ix < 0 ? 0 : (ix >= W ? W - 1 : ix);
        const float v = in[sbase[sub] + icc1 * HW + iyc * W + ixc];
        const bool ok = kok1 && (iy == iyc) && (ix == ixc);
        pack |= (unsigned int)__builtin_bit_cast(unsigned short,
                                                 f2bf(ok ? v : 0.f)) << 16;
      }
      *(unsigned int*)&ldsB[buf][sub * 16 + n_s][kl] = pack;
    }
    __syncthreads();

    // ---- A (weights): unconditional clamped loads; B zeros kill the tail --
    v16bf A;
    #pragma unroll
    for (int i = 0; i < 8; ++i) {
      const int kk = khiA + 2 * i;
      const int k  = k0 + kk;
      const int kc = (k + 1 < K) ? k : (K - 2);
      A[2 * i]     = f2bf(wrow[kc]);
      A[2 * i + 1] = f2bf(wrow[kc + 1]);
    }

    // ---- B from LDS: 32 contiguous bytes per lane -------------------------
    const v16bf Bm = *(const v16bf*)&ldsB[buf][colb + n][half * 16];

    acc = __builtin_amdgcn_wmma_f32_16x16x32_bf16(
        false, A, false, Bm, (short)0, acc, false, false);

    // ---- incremental k-decode update (k += 32 = 3*9 + 5) ------------------
    rr0 += 5; ic0 += 3; if (rr0 >= 9) { rr0 -= 9; ++ic0; }
    rr1 += 5; ic1 += 3; if (rr1 >= 9) { rr1 -= 9; ++ic1; }
  }

  // ---- Epilogue: D layout -> row m = r + 8*half, col n --------------------
  if (s < BHW) {
    #pragma unroll
    for (int r = 0; r < 8; ++r) {
      const int oc = oc_base + 8 * half + r;
      if (oc < Cout) {
        float v = acc[r] + bias[oc];
        const long long oidx =
            ((long long)(b * Cout + oc) * HW) + (long long)y * W + x;
        if (res) v += res[oidx];
        if (leaky) v = (v >= 0.f) ? v : 0.1f * v;
        out[oidx] = v;
      }
    }
  }
}

// ---------------------------------------------------------------------------
// Correlation volume: out[b, 9*(di+4)+(dj+4), y, x] = mean_c f1*shift(f2).
// ---------------------------------------------------------------------------
__global__ void corr_kernel(const float* __restrict__ f1,
                            const float* __restrict__ f2,
                            float* __restrict__ out,
                            int NB, int C, int H, int W)
{
  const int HW = H * W;
  const long long total = (long long)NB * 81 * HW;
  const long long idx = (long long)blockIdx.x * blockDim.x + threadIdx.x;
  if (idx >= total) return;
  int x = (int)(idx % W);
  long long t = idx / W;
  int y = (int)(t % H); t /= H;
  int d = (int)(t % 81);
  int b = (int)(t / 81);
  const int di = d / 9 - 4, dj = d % 9 - 4;
  const int y2 = y + di, x2 = x + dj;
  float s = 0.f;
  if (y2 >= 0 && y2 < H && x2 >= 0 && x2 < W) {
    const float* p1 = f1 + (long long)b * C * HW + y * W + x;
    const float* p2 = f2 + (long long)b * C * HW + y2 * W + x2;
    for (int c = 0; c < C; ++c) s += p1[(long long)c * HW] * p2[(long long)c * HW];
  }
  out[idx] = s / (float)C;
}

// ---------------------------------------------------------------------------
// Bilinear warp with per-tap validity masking (matches reference semantics).
// ---------------------------------------------------------------------------
__global__ void warp_kernel(const float* __restrict__ src,
                            const float* __restrict__ flow,
                            float* __restrict__ out,
                            int NB, int C, int H, int W)
{
  const int HW = H * W;
  const long long total = (long long)NB * C * HW;
  const long long idx = (long long)blockIdx.x * blockDim.x + threadIdx.x;
  if (idx >= total) return;
  int x = (int)(idx % W);
  long long t = idx / W;
  int y = (int)(t % H); t /= H;
  int c = (int)(t % C);
  int b = (int)(t / C);

  const float fx = flow[((long long)b * 2 + 0) * HW + y * W + x];
  const float fy = flow[((long long)b * 2 + 1) * HW + y * W + x];
  const float xsf = (float)x + fx;
  const float ysf = (float)y + fy;
  const float x0f = floorf(xsf), y0f = floorf(ysf);
  const float wx = xsf - x0f, wy = ysf - y0f;
  const int x0 = (int)x0f, y0 = (int)y0f;

  const float* plane = src + ((long long)b * C + c) * HW;
  float acc = 0.f;
  #pragma unroll
  for (int ty = 0; ty < 2; ++ty) {
    #pragma unroll
    for (int tx = 0; tx < 2; ++tx) {
      const int yi = y0 + ty, xi = x0 + tx;
      const bool valid = (xi >= 0) && (xi <= W - 1) && (yi >= 0) && (yi <= H - 1);
      const int yc = yi < 0 ? 0 : (yi > H - 1 ? H - 1 : yi);
      const int xc = xi < 0 ? 0 : (xi > W - 1 ? W - 1 : xi);
      const float v = plane[(long long)yc * W + xc];
      const float wgt = (ty ? wy : 1.f - wy) * (tx ? wx : 1.f - wx);
      acc += (valid ? v : 0.f) * wgt;
    }
  }
  out[idx] = acc;
}

// ---------------------------------------------------------------------------
// Half-pixel bilinear resize with fused scalar multiply (up2 *2, final *4).
// ---------------------------------------------------------------------------
__global__ void resize_kernel(const float* __restrict__ in, float* __restrict__ out,
                              int NB, int C, int Hi, int Wi, int Ho, int Wo,
                              float mul)
{
  const long long total = (long long)NB * C * Ho * Wo;
  const long long idx = (long long)blockIdx.x * blockDim.x + threadIdx.x;
  if (idx >= total) return;
  int x = (int)(idx % Wo);
  long long t = idx / Wo;
  int y = (int)(t % Ho); t /= Ho;
  int c = (int)(t % C);
  int b = (int)(t / C);

  const float sy = (float)Hi / (float)Ho;
  const float sx = (float)Wi / (float)Wo;
  float fy = ((float)y + 0.5f) * sy - 0.5f;
  float fx = ((float)x + 0.5f) * sx - 0.5f;
  fy = fy < 0.f ? 0.f : (fy > (float)(Hi - 1) ? (float)(Hi - 1) : fy);
  fx = fx < 0.f ? 0.f : (fx > (float)(Wi - 1) ? (float)(Wi - 1) : fx);
  const int y0 = (int)fy, x0 = (int)fx;
  const int y1 = y0 + 1 < Hi ? y0 + 1 : Hi - 1;
  const int x1 = x0 + 1 < Wi ? x0 + 1 : Wi - 1;
  const float wy = fy - (float)y0, wx = fx - (float)x0;

  const float* p = in + ((long long)b * C + c) * Hi * Wi;
  const float v00 = p[(long long)y0 * Wi + x0];
  const float v01 = p[(long long)y0 * Wi + x1];
  const float v10 = p[(long long)y1 * Wi + x0];
  const float v11 = p[(long long)y1 * Wi + x1];
  const float v = (v00 * (1.f - wx) + v01 * wx) * (1.f - wy) +
                  (v10 * (1.f - wx) + v11 * wx) * wy;
  out[idx] = v * mul;
}

// ---------------------------------------------------------------------------
// Channel concat of up to 3 tensors (pass Cc = 0 for 2-way).
// ---------------------------------------------------------------------------
__global__ void cat3_kernel(const float* __restrict__ a, int Ca,
                            const float* __restrict__ b, int Cb,
                            const float* __restrict__ c, int Cc,
                            float* __restrict__ out, int NB, int HW)
{
  const int Ct = Ca + Cb + Cc;
  const long long total = (long long)NB * Ct * HW;
  const long long idx = (long long)blockIdx.x * blockDim.x + threadIdx.x;
  if (idx >= total) return;
  int p = (int)(idx % HW);
  long long t = idx / HW;
  int ch = (int)(t % Ct);
  int n  = (int)(t / Ct);
  float v;
  if (ch < Ca)            v = a[((long long)n * Ca + ch) * HW + p];
  else if (ch < Ca + Cb)  v = b[((long long)n * Cb + (ch - Ca)) * HW + p];
  else                    v = c[((long long)n * Cc + (ch - Ca - Cb)) * HW + p];
  out[idx] = v;
}

// ---------------------------------------------------------------------------
// Host-side orchestration
// ---------------------------------------------------------------------------
namespace {

constexpr int NB = 4;

inline int cdiv(long long a, long long b) { return (int)((a + b - 1) / b); }

inline void launch_conv(hipStream_t st, const float* in, const float* w,
                        const float* bias, const float* res, float* out,
                        int Cin, int H, int W, int Cout, int dil, int leaky) {
  const long long BHW = (long long)NB * H * W;
  if (Cout <= 2) {
    conv3x3_wmma_bf16<8><<<cdiv(BHW, 128), 256, 0, st>>>(
        in, w, bias, res, out, NB, Cin, H, W, Cout, dil, leaky);
  } else if (Cout <= 32) {
    conv3x3_wmma_bf16<4><<<cdiv(BHW, 64), 256, 0, st>>>(
        in, w, bias, res, out, NB, Cin, H, W, Cout, dil, leaky);
  } else if (Cout <= 64) {
    conv3x3_wmma_bf16<2><<<cdiv(BHW, 32), 256, 0, st>>>(
        in, w, bias, res, out, NB, Cin, H, W, Cout, dil, leaky);
  } else {
    conv3x3_wmma_bf16<1><<<cdiv(BHW, 16), 256, 0, st>>>(
        in, w, bias, res, out, NB, Cin, H, W, Cout, dil, leaky);
  }
}

inline void launch_corr(hipStream_t st, const float* f1, const float* f2,
                        float* out, int C, int H, int W) {
  const long long n = (long long)NB * 81 * H * W;
  corr_kernel<<<cdiv(n, 256), 256, 0, st>>>(f1, f2, out, NB, C, H, W);
}

inline void launch_warp(hipStream_t st, const float* src, const float* flow,
                        float* out, int C, int H, int W) {
  const long long n = (long long)NB * C * H * W;
  warp_kernel<<<cdiv(n, 256), 256, 0, st>>>(src, flow, out, NB, C, H, W);
}

inline void launch_resize(hipStream_t st, const float* in, float* out, int C,
                          int Hi, int Wi, int Ho, int Wo, float mul) {
  const long long n = (long long)NB * C * Ho * Wo;
  resize_kernel<<<cdiv(n, 256), 256, 0, st>>>(in, out, NB, C, Hi, Wi, Ho, Wo, mul);
}

inline void launch_cat3(hipStream_t st, const float* a, int Ca, const float* b,
                        int Cb, const float* c, int Cc, float* out, int HW) {
  const long long n = (long long)NB * (Ca + Cb + Cc) * HW;
  cat3_kernel<<<cdiv(n, 256), 256, 0, st>>>(a, Ca, b, Cb, c, Cc, out, NB, HW);
}

struct Bufs {
  float *warpb, *corrb, *catb, *x0, *x1, *x2, *x3, *x4;
};

// Param leaf order (jax.tree sorted keys): per conv {b, w}.
inline const float* Pw(void* const* din, int base, int ci) {
  return (const float*)din[base + 2 * ci + 1];
}
inline const float* Pb(void* const* din, int base, int ci) {
  return (const float*)din[base + 2 * ci];
}

// Decoder: c0..c4 + pf (keys sorted: c0,c1,c2,c3,c4,pf -> indices 0..5).
void run_decoder(hipStream_t st, void* const* din, int base, const float* in,
                 int Cin, int H, int W, const Bufs& B, float* flow,
                 const float* res) {
  const int HW = H * W;
  launch_conv(st, in,   Pw(din, base, 0), Pb(din, base, 0), nullptr, B.x0, Cin, H, W, 128, 1, 1);
  launch_conv(st, B.x0, Pw(din, base, 1), Pb(din, base, 1), nullptr, B.x1, 128, H, W, 128, 1, 1);
  launch_cat3(st, B.x0, 128, B.x1, 128, nullptr, 0, B.catb, HW);
  launch_conv(st, B.catb, Pw(din, base, 2), Pb(din, base, 2), nullptr, B.x2, 256, H, W, 96, 1, 1);
  launch_cat3(st, B.x1, 128, B.x2, 96, nullptr, 0, B.catb, HW);
  launch_conv(st, B.catb, Pw(din, base, 3), Pb(din, base, 3), nullptr, B.x3, 224, H, W, 64, 1, 1);
  launch_cat3(st, B.x2, 96, B.x3, 64, nullptr, 0, B.catb, HW);
  launch_conv(st, B.catb, Pw(din, base, 4), Pb(din, base, 4), nullptr, B.x4, 160, H, W, 32, 1, 1);
  launch_cat3(st, B.x3, 64, B.x4, 32, nullptr, 0, B.catb, HW);
  launch_conv(st, B.catb, Pw(din, base, 5), Pb(din, base, 5), res, flow, 96, H, W, 2, 1, 0);
}

void run_level(hipStream_t st, void* const* din, int base, const float* c1f,
               const float* c2f, int Cf, const float* up, int H, int W,
               const Bufs& B, float* flow) {
  launch_warp(st, c2f, up, B.warpb, Cf, H, W);
  launch_corr(st, c1f, B.warpb, B.corrb, Cf, H, W);
  launch_cat3(st, B.corrb, 81, c1f, Cf, up, 2, B.catb, H * W);
  run_decoder(st, din, base, B.catb, 81 + Cf + 2, H, W, B, flow, up);
}

} // namespace

extern "C" void kernel_launch(void* const* d_in, const int* in_sizes, int n_in,
                              void* d_out, int out_size, void* d_ws, size_t ws_size,
                              hipStream_t stream) {
  (void)in_sizes; (void)n_in; (void)out_size; (void)ws_size;

  // Static shapes (B=4, img 256x832). Level l: H=256/2^l, W=832/2^l.
  const int H6 = 4,  W6 = 13;
  const int H5 = 8,  W5 = 26;
  const int H4 = 16, W4 = 52;
  const int H3 = 32, W3 = 104;
  const int H2 = 64, W2 = 208;

  // Feature pyramids (tree-leaf order): c11..c16 = d_in[0..5], c21..c26 = [6..11].
  const float* c12 = (const float*)d_in[1];
  const float* c13 = (const float*)d_in[2];
  const float* c14 = (const float*)d_in[3];
  const float* c15 = (const float*)d_in[4];
  const float* c16 = (const float*)d_in[5];
  const float* c22 = (const float*)d_in[7];
  const float* c23 = (const float*)d_in[8];
  const float* c24 = (const float*)d_in[9];
  const float* c25 = (const float*)d_in[10];
  const float* c26 = (const float*)d_in[11];
  // d_in[12]=img_h, d_in[13]=img_w (static). Params (sorted keys) from 14:
  const int DC = 14;   // dc: c1..c7, 14 leaves
  const int L2 = 28, L3 = 40, L4 = 52, L5 = 64, L6 = 76;  // 12 leaves each

  // ---- workspace bump layout (fp32), sized for the level-2 maxima ---------
  float* ws = (float*)d_ws;
  size_t off = 0;
  auto alloc = [&](size_t n) { float* p = ws + off; off += (n + 63) & ~(size_t)63; return p; };
  const size_t SP2 = (size_t)NB * H2 * W2;                 // 53248 * 4
  Bufs B;
  B.warpb = alloc(128 * (size_t)NB * H5 * W5 > 32 * SP2 ? 128 * (size_t)NB * H5 * W5 : 32 * SP2);
  B.corrb = alloc(81 * SP2);
  B.catb  = alloc(256 * SP2);
  B.x0    = alloc(128 * SP2);
  B.x1    = alloc(128 * SP2);
  B.x2    = alloc(96 * SP2);
  B.x3    = alloc(64 * SP2);
  B.x4    = alloc(32 * SP2);
  float* up     = alloc(2 * SP2);
  float* flow6  = alloc(2 * (size_t)NB * H6 * W6);
  float* flow5  = alloc(2 * (size_t)NB * H5 * W5);
  float* flow4  = alloc(2 * (size_t)NB * H4 * W4);
  float* flow3  = alloc(2 * (size_t)NB * H3 * W3);
  float* flow2  = alloc(2 * SP2);
  float* flow2b = alloc(2 * SP2);

  // ---- level 6 ------------------------------------------------------------
  launch_corr(stream, c16, c26, B.corrb, 196, H6, W6);
  run_decoder(stream, d_in, L6, B.corrb, 81, H6, W6, B, flow6, nullptr);
  launch_resize(stream, flow6, up, 2, H6, W6, H5, W5, 2.0f);

  // ---- levels 5..2 --------------------------------------------------------
  run_level(stream, d_in, L5, c15, c25, 128, up, H5, W5, B, flow5);
  launch_resize(stream, flow5, up, 2, H5, W5, H4, W4, 2.0f);
  run_level(stream, d_in, L4, c14, c24, 96, up, H4, W4, B, flow4);
  launch_resize(stream, flow4, up, 2, H4, W4, H3, W3, 2.0f);
  run_level(stream, d_in, L3, c13, c23, 64, up, H3, W3, B, flow3);
  launch_resize(stream, flow3, up, 2, H3, W3, H2, W2, 2.0f);
  run_level(stream, d_in, L2, c12, c22, 32, up, H2, W2, B, flow2);

  // ---- context network (dilated convs) ------------------------------------
  launch_cat3(stream, flow2, 2, B.x4, 32, nullptr, 0, B.catb, H2 * W2);
  launch_conv(stream, B.catb, Pw(d_in, DC, 0), Pb(d_in, DC, 0), nullptr, B.x0, 34,  H2, W2, 128, 1,  1);
  launch_conv(stream, B.x0,   Pw(d_in, DC, 1), Pb(d_in, DC, 1), nullptr, B.x1, 128, H2, W2, 128, 2,  1);
  launch_conv(stream, B.x1,   Pw(d_in, DC, 2), Pb(d_in, DC, 2), nullptr, B.x0, 128, H2, W2, 128, 4,  1);
  launch_conv(stream, B.x0,   Pw(d_in, DC, 3), Pb(d_in, DC, 3), nullptr, B.x2, 128, H2, W2, 96,  8,  1);
  launch_conv(stream, B.x2,   Pw(d_in, DC, 4), Pb(d_in, DC, 4), nullptr, B.x3, 96,  H2, W2, 64,  16, 1);
  launch_conv(stream, B.x3,   Pw(d_in, DC, 5), Pb(d_in, DC, 5), nullptr, B.x4, 64,  H2, W2, 32,  1,  1);
  launch_conv(stream, B.x4,   Pw(d_in, DC, 6), Pb(d_in, DC, 6), flow2,   flow2b, 32, H2, W2, 2,  1,  0);

  // ---- final bilinear upsamples (*4) into concatenated d_out --------------
  float* out = (float*)d_out;
  size_t o = 0;
  launch_resize(stream, flow2b, out + o, 2, H2, W2, 256, 832, 4.0f); o += (size_t)NB * 2 * 256 * 832;
  launch_resize(stream, flow3,  out + o, 2, H3, W3, 128, 416, 4.0f); o += (size_t)NB * 2 * 128 * 416;
  launch_resize(stream, flow4,  out + o, 2, H4, W4, 64,  208, 4.0f); o += (size_t)NB * 2 * 64 * 208;
  launch_resize(stream, flow5,  out + o, 2, H5, W5, 32,  104, 4.0f);
}
